// TransformerDecoder_40793599377934
// MI455X (gfx1250) — compile-verified, matrix operand order checked by reading, not run on hardware
//
#include <hip/hip_runtime.h>

// ---------------- problem constants ----------------
constexpr int SQ = 2048;   // seq length
constexpr int DM = 1024;   // model dim
constexpr int NH = 16;     // heads
constexpr int HD = 64;     // head dim
constexpr int NL = 4;      // layers
constexpr int VO = 32000;  // vocab

// ---------------- types ----------------
typedef __bf16 bf16_t;
typedef __attribute__((ext_vector_type(8)))  __bf16 v8bf;
typedef __attribute__((ext_vector_type(16))) __bf16 v16bf;
typedef __attribute__((ext_vector_type(8)))  float  v8f;
typedef __attribute__((ext_vector_type(8)))  float  f32x8;
typedef __attribute__((ext_vector_type(4)))  float  f32x4;
typedef __attribute__((address_space(3))) char lds_char_t;

__device__ __forceinline__ unsigned short f2bf(float x) {
    unsigned u = __float_as_uint(x);
    return (unsigned short)((u + 0x7FFFu + ((u >> 16) & 1u)) >> 16);
}

// ---------------- CDNA5 async global->LDS copy (16B per lane) ----------------
__device__ __forceinline__ void async_copy16(const void* g, void* l) {
    // 32-bit LDS byte address (widen first so the host pass accepts the cast)
    unsigned laddr = (unsigned)(unsigned long long)(lds_char_t*)l;
    asm volatile("global_load_async_to_lds_b128 %0, %1, off"
                 :: "v"(laddr), "v"(g) : "memory");
}
__device__ __forceinline__ void wait_async_le8() {
    asm volatile("s_wait_asynccnt 8" ::: "memory");
}
__device__ __forceinline__ void wait_async_le0() {
    asm volatile("s_wait_asynccnt 0" ::: "memory");
}

// ---------------- LDS tile geometry ----------------
// A tile: 256 rows x 32 K (bf16). row stride 64B, 4 granules of 16B (8 bf16).
//   granule (r,g) at: r*64 + ((g ^ ((r>>2)&3)) << 4)        -> conflict-free frag reads
// B tile: 128 cols x 32 K (fp32). col stride 128B, 8 granules of 16B (4 f32).
//   granule (c,g) at: c*128 + ((g ^ ((c>>1)&7)) << 4)       -> conflict-free frag reads
constexpr int LDS_A_BYTES = 256 * 64;    // 16 KB
constexpr int LDS_B_BYTES = 128 * 128;   // 16 KB
constexpr int LDS_BUF     = LDS_A_BYTES + LDS_B_BYTES;  // 32 KB, x2 buffers

// issue one block-tile fill: 8 async b128 per thread (4 for A, 4 for B)
__device__ __forceinline__ void fill_tile(const bf16_t* A, int lda,
                                          const float* B, int ldb,
                                          int row0b, int col0b, int N, int k0,
                                          char* sA, char* sB, int tid) {
    #pragma unroll
    for (int i = 0; i < 4; ++i) {               // A: 1024 granules
        int gid = tid + i * 256;
        int r = gid >> 2, g = gid & 3;
        const void* gp = (const void*)(A + (size_t)(row0b + r) * lda + k0 + g * 8);
        void* lp = sA + r * 64 + ((g ^ ((r >> 2) & 3)) << 4);
        async_copy16(gp, lp);
    }
    #pragma unroll
    for (int i = 0; i < 4; ++i) {               // B: 1024 granules
        int gid = tid + i * 256;
        int c = gid >> 3, g = gid & 7;
        int cg = col0b + c; if (cg > N - 1) cg = N - 1;   // clamp (N=64 case)
        const void* gp = (const void*)(B + (size_t)cg * ldb + k0 + g * 4);
        void* lp = sB + c * 128 + ((g ^ ((c >> 1) & 7)) << 4);
        async_copy16(gp, lp);
    }
}

// ---- A fragment from LDS (bf16 16x32), ISA 7.12.2 16-bit A layout ----
__device__ __forceinline__ v16bf frag_a_lds(const char* sA, int rbase, int lane) {
    int r  = rbase + (lane & 15);
    int g0 = (lane >> 4) ? 1 : 0;               // kb=8 -> granule 1
    const char* rowp = sA + r * 64;
    int sw = (r >> 2) & 3;
    v8bf lo = *(const v8bf*)(rowp + (((g0 + 0) ^ sw) << 4));   // K [kb..kb+7]
    v8bf hi = *(const v8bf*)(rowp + (((g0 + 2) ^ sw) << 4));   // K [kb+16..kb+23]
    return __builtin_shufflevector(lo, hi, 0,1,2,3,4,5,6,7,8,9,10,11,12,13,14,15);
}

// ---- B fragment from LDS (fp32 -> bf16 in-register) ----
__device__ __forceinline__ v16bf frag_b_lds(const char* sB, int cbase, int lane) {
    int c  = cbase + (lane & 15);
    int g0 = (lane >> 4) ? 2 : 0;               // kb=8 -> floats 8..15 = granules 2,3
    const char* rowp = sB + c * 128;
    int sw = (c >> 1) & 7;
    f32x4 q0 = *(const f32x4*)(rowp + (((g0 + 0) ^ sw) << 4));
    f32x4 q1 = *(const f32x4*)(rowp + (((g0 + 1) ^ sw) << 4));
    f32x4 q2 = *(const f32x4*)(rowp + (((g0 + 4) ^ sw) << 4));
    f32x4 q3 = *(const f32x4*)(rowp + (((g0 + 5) ^ sw) << 4));
    f32x8 lo = __builtin_shufflevector(q0, q1, 0,1,2,3,4,5,6,7);
    f32x8 hi = __builtin_shufflevector(q2, q3, 0,1,2,3,4,5,6,7);
    v8bf ab = __builtin_convertvector(lo, v8bf);
    v8bf bb = __builtin_convertvector(hi, v8bf);
    return __builtin_shufflevector(ab, bb, 0,1,2,3,4,5,6,7,8,9,10,11,12,13,14,15);
}

// ---------------- LDS-staged, double-buffered TN WMMA GEMM ----------------
// C[M,N] = alpha * (A[M,K](bf16) @ B[N,K](f32)^T) + bias[N]  (+optional relu)
// block = 256 threads = 8 waves; wave -> 64x64 tile; block tile 256(M) x 128(N)
__global__ __launch_bounds__(256)
void gemm_bf16_k(const unsigned short* __restrict__ Au, int lda,
                 const float* __restrict__ B, int ldb,
                 const float* __restrict__ bias,
                 float* __restrict__ C, int ldc,
                 int M, int N, int K, float alpha, int relu) {
    __shared__ char smem[2 * LDS_BUF];          // 64 KB
    const bf16_t* A = (const bf16_t*)Au;
    int tid  = threadIdx.x;
    int wave = tid >> 5;
    int lane = tid & 31;
    int row0b = blockIdx.y * 256;
    int col0b = blockIdx.x * 128;
    int row0  = row0b + (wave & 3) * 64;
    int col0  = col0b + (wave >> 2) * 64;
    bool do_store = (row0 < M) && (col0 < N);
    // clamp so out-of-range waves still do (redundant) work and hit all barriers
    int row0c = row0 < M - 64 ? row0 : M - 64;
    int col0c = col0 < N - 64 ? col0 : (N - 64 > 0 ? N - 64 : 0);
    int rbase = row0c - row0b;                  // relative tile coords in LDS
    int cbase = col0c - col0b;

    v8f acc[4][4] = {};
    fill_tile(A, lda, B, ldb, row0b, col0b, N, 0, smem, smem + LDS_A_BYTES, tid);
    int par = 0;
    for (int k0 = 0; k0 < K; k0 += 32) {
        bool more = (k0 + 32) < K;
        if (more) {
            int nxt = par ^ 1;
            fill_tile(A, lda, B, ldb, row0b, col0b, N, k0 + 32,
                      smem + nxt * LDS_BUF, smem + nxt * LDS_BUF + LDS_A_BYTES, tid);
            wait_async_le8();                   // current buffer's 8 asyncs done
        } else {
            wait_async_le0();
        }
        __syncthreads();                        // tile visible to all waves
        const char* sA = smem + par * LDS_BUF;
        const char* sB = smem + par * LDS_BUF + LDS_A_BYTES;
        v16bf af[4], bfr[4];
        #pragma unroll
        for (int i = 0; i < 4; ++i) af[i]  = frag_a_lds(sA, rbase + 16 * i, lane);
        #pragma unroll
        for (int j = 0; j < 4; ++j) bfr[j] = frag_b_lds(sB, cbase + 16 * j, lane);
        #pragma unroll
        for (int i = 0; i < 4; ++i)
            #pragma unroll
            for (int j = 0; j < 4; ++j)
                acc[i][j] = __builtin_amdgcn_wmma_f32_16x16x32_bf16(
                    false, af[i], false, bfr[j], (short)0, acc[i][j], false, false);
        __syncthreads();                        // all reads done before refill
        par ^= 1;
    }

    if (!do_store) return;
    // C/D layout: VGPR r, lanes 0-15 -> (M=r, N=lane); lanes 16-31 -> (M=8+r, N=lane-16)
    int rb  = (lane >> 4) ? 8 : 0;
    int ccl = lane & 15;
    #pragma unroll
    for (int j = 0; j < 4; ++j) {
        int col = col0c + 16 * j + ccl;
        float bv = bias ? bias[col] : 0.0f;
        #pragma unroll
        for (int i = 0; i < 4; ++i) {
            #pragma unroll
            for (int r = 0; r < 8; ++r) {
                int row = row0c + 16 * i + rb + r;
                float v = alpha * acc[i][j][r] + bv;
                if (relu) v = v > 0.0f ? v : 0.0f;
                C[(size_t)row * ldc + col] = v;
            }
        }
    }
}

// ---------------- embedding + positional ----------------
__global__ void embed_k(const int* __restrict__ ids, const float* __restrict__ emb,
                        const float* __restrict__ pos, float* __restrict__ h) {
    int s = blockIdx.x;
    int id = ids[s];
    for (int d = threadIdx.x; d < DM; d += blockDim.x)
        h[(size_t)s * DM + d] = emb[(size_t)id * DM + d] + pos[(size_t)s * DM + d];
}

// ---------------- fp32 -> bf16 (RNE) ----------------
__global__ void cvt_bf16_k(const float* __restrict__ in, unsigned short* __restrict__ out, int n) {
    int i = blockIdx.x * blockDim.x + threadIdx.x;
    if (i < n) out[i] = f2bf(in[i]);
}

// ---------------- causal softmax: fp32 scores -> bf16 probs ----------------
__global__ __launch_bounds__(256)
void softmax_k(const float* __restrict__ sc, unsigned short* __restrict__ pb) {
    __shared__ float red[256];
    int q = blockIdx.x, t = threadIdx.x;
    const float* srow = sc + (size_t)q * SQ;
    unsigned short* prow = pb + (size_t)q * SQ;
    float mx = -3.402823466e38f;
    for (int c = t; c <= q; c += 256) mx = fmaxf(mx, srow[c]);
    red[t] = mx; __syncthreads();
    for (int o = 128; o > 0; o >>= 1) { if (t < o) red[t] = fmaxf(red[t], red[t + o]); __syncthreads(); }
    mx = red[0]; __syncthreads();
    float sum = 0.0f;
    for (int c = t; c <= q; c += 256) sum += __expf(srow[c] - mx);
    red[t] = sum; __syncthreads();
    for (int o = 128; o > 0; o >>= 1) { if (t < o) red[t] += red[t + o]; __syncthreads(); }
    float inv = 1.0f / red[0];
    for (int c = t; c < SQ; c += 256) {
        float p = (c <= q) ? __expf(srow[c] - mx) * inv : 0.0f;
        prow[c] = f2bf(p);
    }
}

// ---------------- per-head V transpose: vt[d][s] = v[s][hoff+d] ----------------
__global__ void vtrans_k(const float* __restrict__ v, float* __restrict__ vt, int hoff) {
    int i = blockIdx.x * blockDim.x + threadIdx.x;   // over HD*SQ
    int s = i & (SQ - 1);
    int d = i >> 11;                                  // SQ = 2^11
    vt[(size_t)d * SQ + s] = v[(size_t)s * DM + hoff + d];
}

// ---------------- fused residual add + LayerNorm (in-place on h) ----------------
__global__ __launch_bounds__(256)
void add_ln_k(float* __restrict__ h, const float* __restrict__ dlt,
              const float* __restrict__ g, const float* __restrict__ b) {
    __shared__ float red[256];
    int s = blockIdx.x, t = threadIdx.x;
    float x[4]; float sum = 0.0f;
    #pragma unroll
    for (int i = 0; i < 4; ++i) {
        int d = t + 256 * i;
        x[i] = h[(size_t)s * DM + d] + dlt[(size_t)s * DM + d];
        sum += x[i];
    }
    red[t] = sum; __syncthreads();
    for (int o = 128; o > 0; o >>= 1) { if (t < o) red[t] += red[t + o]; __syncthreads(); }
    float mu = red[0] / DM; __syncthreads();
    float vs = 0.0f;
    #pragma unroll
    for (int i = 0; i < 4; ++i) { float dv = x[i] - mu; vs += dv * dv; }
    red[t] = vs; __syncthreads();
    for (int o = 128; o > 0; o >>= 1) { if (t < o) red[t] += red[t + o]; __syncthreads(); }
    float inv = rsqrtf(red[0] / DM + 1e-5f);
    #pragma unroll
    for (int i = 0; i < 4; ++i) {
        int d = t + 256 * i;
        h[(size_t)s * DM + d] = (x[i] - mu) * inv * g[d] + b[d];
    }
}

// ---------------- host orchestration ----------------
static void launch_gemm(const unsigned short* A, int lda, const float* B, int ldb,
                        const float* bias, float* C, int ldc, int M, int N, int K,
                        float alpha, int relu, hipStream_t st) {
    dim3 grid((N + 127) / 128, (M + 255) / 256);
    gemm_bf16_k<<<grid, dim3(256), 0, st>>>(A, lda, B, ldb, bias, C, ldc, M, N, K, alpha, relu);
}

extern "C" void kernel_launch(void* const* d_in, const int* in_sizes, int n_in,
                              void* d_out, int out_size, void* d_ws, size_t ws_size,
                              hipStream_t stream) {
    const int*   ids  = (const int*)  d_in[0];
    const float* emb  = (const float*)d_in[1];
    const float* pos  = (const float*)d_in[2];
    const float* wq   = (const float*)d_in[3];  const float* bq = (const float*)d_in[4];
    const float* wk   = (const float*)d_in[5];  const float* bk = (const float*)d_in[6];
    const float* wv   = (const float*)d_in[7];  const float* bv = (const float*)d_in[8];
    const float* wo   = (const float*)d_in[9];  const float* bo = (const float*)d_in[10];
    const float* ln1g = (const float*)d_in[11]; const float* ln1b = (const float*)d_in[12];
    const float* w1   = (const float*)d_in[13]; const float* b1 = (const float*)d_in[14];
    const float* w2   = (const float*)d_in[15]; const float* b2 = (const float*)d_in[16];
    const float* ln2g = (const float*)d_in[17]; const float* ln2b = (const float*)d_in[18];
    const float* lmh  = (const float*)d_in[19];
    float* out = (float*)d_out;

    // workspace carve-up (~130 MB)
    char* w = (char*)d_ws;
    float*          h    = (float*)w;          w += (size_t)SQ * DM * 4;       // residual stream
    unsigned short* abf  = (unsigned short*)w; w += (size_t)SQ * 4 * DM * 2;   // generic bf16 A buffer
    float*          qb   = (float*)w;          w += (size_t)SQ * DM * 4;
    float*          kb   = (float*)w;          w += (size_t)SQ * DM * 4;
    float*          vb   = (float*)w;          w += (size_t)SQ * DM * 4;
    float*          ob   = (float*)w;          w += (size_t)SQ * DM * 4;       // attn output
    float*          m1   = (float*)w;          w += (size_t)SQ * 4 * DM * 4;   // FFN hidden
    float*          sc   = (float*)w;          w += (size_t)SQ * SQ * 4;       // one head's scores
    unsigned short* pb   = (unsigned short*)w; w += (size_t)SQ * SQ * 2;       // one head's probs (bf16)
    float*          vt   = (float*)w;          w += (size_t)HD * SQ * 4;       // transposed V head
    float*          tb   = (float*)w;          w += (size_t)SQ * DM * 4;       // proj/FFN result before LN
    (void)ws_size; (void)in_sizes; (void)n_in; (void)out_size;

    const int CVT = 256;
    const float inv_sqrt_hd = 0.125f;   // 1/sqrt(64)

    // h = embed[ids] + pos
    embed_k<<<SQ, 256, 0, stream>>>(ids, emb, pos, h);

    for (int l = 0; l < NL; ++l) {
        const float* lwq = wq + (size_t)l * DM * DM;  const float* lbq = bq + (size_t)l * DM;
        const float* lwk = wk + (size_t)l * DM * DM;  const float* lbk = bk + (size_t)l * DM;
        const float* lwv = wv + (size_t)l * DM * DM;  const float* lbv = bv + (size_t)l * DM;
        const float* lwo = wo + (size_t)l * DM * DM;  const float* lbo = bo + (size_t)l * DM;
        const float* lw1 = w1 + (size_t)l * 4 * DM * DM; const float* lb1 = b1 + (size_t)l * 4 * DM;
        const float* lw2 = w2 + (size_t)l * DM * 4 * DM; const float* lb2 = b2 + (size_t)l * DM;

        // --- QKV projections ---
        int n = SQ * DM;
        cvt_bf16_k<<<(n + CVT - 1) / CVT, CVT, 0, stream>>>(h, abf, n);
        launch_gemm(abf, DM, lwq, DM, lbq, qb, DM, SQ, DM, DM, 1.0f, 0, stream);
        launch_gemm(abf, DM, lwk, DM, lbk, kb, DM, SQ, DM, DM, 1.0f, 0, stream);
        launch_gemm(abf, DM, lwv, DM, lbv, vb, DM, SQ, DM, DM, 1.0f, 0, stream);

        // --- attention, head by head ---
        cvt_bf16_k<<<(n + CVT - 1) / CVT, CVT, 0, stream>>>(qb, abf, n);  // Q in bf16
        for (int hh = 0; hh < NH; ++hh) {
            int hoff = hh * HD;
            // scores = (Q_h @ K_h^T) / sqrt(HD)   [SQ,SQ], K-dim = 64
            launch_gemm(abf + hoff, DM, kb + hoff, DM, nullptr,
                        sc, SQ, SQ, SQ, HD, inv_sqrt_hd, 0, stream);
            // causal softmax -> bf16 probs
            softmax_k<<<SQ, 256, 0, stream>>>(sc, pb);
            // transpose V head -> vt[HD,SQ]
            vtrans_k<<<(HD * SQ) / 256, 256, 0, stream>>>(vb, vt, hoff);
            // O_h = P @ V_h    [SQ,HD], K-dim = SQ
            launch_gemm(pb, SQ, vt, SQ, nullptr,
                        ob + hoff, DM, SQ, HD, SQ, 1.0f, 0, stream);
        }

        // --- output projection + residual + LN1 ---
        cvt_bf16_k<<<(n + CVT - 1) / CVT, CVT, 0, stream>>>(ob, abf, n);
        launch_gemm(abf, DM, lwo, DM, lbo, tb, DM, SQ, DM, DM, 1.0f, 0, stream);
        add_ln_k<<<SQ, 256, 0, stream>>>(h, tb, ln1g + (size_t)l * DM, ln1b + (size_t)l * DM);

        // --- FFN ---
        cvt_bf16_k<<<(n + CVT - 1) / CVT, CVT, 0, stream>>>(h, abf, n);
        launch_gemm(abf, DM, lw1, DM, lb1, m1, 4 * DM, SQ, 4 * DM, DM, 1.0f, 1, stream);  // relu
        int n4 = SQ * 4 * DM;
        cvt_bf16_k<<<(n4 + CVT - 1) / CVT, CVT, 0, stream>>>(m1, abf, n4);
        launch_gemm(abf, 4 * DM, lw2, 4 * DM, lb2, tb, DM, SQ, DM, 4 * DM, 1.0f, 0, stream);
        add_ln_k<<<SQ, 256, 0, stream>>>(h, tb, ln2g + (size_t)l * DM, ln2b + (size_t)l * DM);
    }

    // --- LM head: logits[SQ, VO] ---
    int n = SQ * DM;
    cvt_bf16_k<<<(n + CVT - 1) / CVT, CVT, 0, stream>>>(h, abf, n);
    launch_gemm(abf, DM, lmh, DM, nullptr, out, VO, SQ, VO, DM, 1.0f, 0, stream);
}
